// A3TGCN_78769700208933
// MI455X (gfx1250) — matrix operand order repaired and checked by reading, not compiled, not run on hardware
//
#include <hip/hip_runtime.h>
#include <hip/hip_bf16.h>

typedef __attribute__((ext_vector_type(16))) _Float16 v16h;
typedef __attribute__((ext_vector_type(8)))  float    v8f;

#define NB_B    2
#define NB_N    50000
#define NB_FIN  2
#define NB_FOUT 32
#define NB_T    12
#define TILES_PER_B (NB_N / 16)      // 3125
#define GRU_WAVES   10               // 6250 tiles / 10 waves = 625 blocks, exact coverage

__device__ __forceinline__ float sigmoidf_(float x) { return 1.0f / (1.0f + __expf(-x)); }

__device__ __forceinline__ void atomic_add_f32(float* p, float v) {
  unsafeAtomicAdd(p, v);   // native global_atomic_add_f32 on CDNA
}

// ---------------- utility ----------------
__global__ void k_zero(float* p, long n) {
  long i = (long)blockIdx.x * blockDim.x + threadIdx.x;
  long s = (long)gridDim.x * blockDim.x;
  for (; i < n; i += s) p[i] = 0.0f;
}

__global__ void k_deg(const int* col, const float* w, float* deg, int E) {
  int e = blockIdx.x * blockDim.x + threadIdx.x;
  if (e < E) atomic_add_f32(&deg[col[e]], w[e]);
}

// dinv = rsqrt(deg + 1)   (self-loop weight folded in; strictly > 0)
__global__ void k_dinv(const float* deg, float* dinv) {
  int i = blockIdx.x * blockDim.x + threadIdx.x;
  if (i < NB_N) dinv[i] = rsqrtf(deg[i] + 1.0f);
}

// norm[e] = dinv[row]*w*dinv[col];  norm[E+i] = dinv[i]^2 (self loops)
__global__ void k_norm(const int* row, const int* col, const float* w,
                       const float* dinv, float* norm, int E) {
  int i = blockIdx.x * blockDim.x + threadIdx.x;
  if (i < E) {
    norm[i] = dinv[row[i]] * w[i] * dinv[col[i]];
  } else if (i < E + NB_N) {
    int n = i - E;
    norm[i] = dinv[n] * dinv[n];
  }
}

__global__ void k_softmax(const float* att, float* probs) {
  if (blockIdx.x == 0 && threadIdx.x == 0) {
    float m = att[0];
    for (int t = 1; t < NB_T; ++t) m = fmaxf(m, att[t]);
    float e[NB_T], s = 0.0f;
    for (int t = 0; t < NB_T; ++t) { e[t] = __expf(att[t] - m); s += e[t]; }
    float inv = 1.0f / s;
    for (int t = 0; t < NB_T; ++t) probs[t] = e[t] * inv;
  }
}

// Fuse the GCN weight into the GRU linear:
//   [gcn_g | H] @ Lg + Lg_b = prop @ (Wg @ Lg_top) + H @ Lg_bot + (bg @ Lg_top + Lg_b)
// C:     [3][2][32]   fused 2x32 weights  (Wg @ Lg_top)
// bfold: [3][32]      fused biases
__global__ void k_fuse(const float* Wz, const float* bz, const float* Wr, const float* br,
                       const float* Wh, const float* bh,
                       const float* Lz, const float* Lzb, const float* Lr, const float* Lrb,
                       const float* Lh, const float* Lhb,
                       float* C, float* bfold) {
  int idx = blockIdx.x * blockDim.x + threadIdx.x;
  if (idx >= 3 * 32) return;
  int o = idx & 31, g = idx >> 5;
  const float* W  = (g == 0) ? Wz : ((g == 1) ? Wr : Wh);
  const float* bg = (g == 0) ? bz : ((g == 1) ? br : bh);
  const float* L  = (g == 0) ? Lz : ((g == 1) ? Lr : Lh);
  const float* Lb = (g == 0) ? Lzb : ((g == 1) ? Lrb : Lhb);
  float c0 = 0.f, c1 = 0.f, bf = Lb[o];
  for (int j = 0; j < NB_FOUT; ++j) {
    float lj = L[j * NB_FOUT + o];       // Lg_top row j
    c0 = fmaf(W[j], lj, c0);             // Wg[0][j]
    c1 = fmaf(W[NB_FOUT + j], lj, c1);   // Wg[1][j]
    bf = fmaf(bg[j], lj, bf);
  }
  C[(g * 2 + 0) * NB_FOUT + o] = c0;
  C[(g * 2 + 1) * NB_FOUT + o] = c1;
  bfold[g * NB_FOUT + o] = bf;
}

// WMMA B-fragments, wB layout: [g(3)][kc(2)][nc(2)][lane(32)][e(16)] halfs.
// 16-bit B (32x16 KxN): elements 0..7 -> K=hi*8+0..7, elements 8..15 -> K=16+hi*8+0..7; col N = lane%16.
// kc=0: rows K=0,1 carry the fused 2x32 weights, K>=2 are zero.
// kc=1: rows 32..63 of Lg_w (the H-part of the GRU linear).
__global__ void k_prepw(const float* Lz, const float* Lr, const float* Lh,
                        const float* C, _Float16* wB) {
  int idx = blockIdx.x * blockDim.x + threadIdx.x;
  if (idx >= 3 * 2 * 2 * 32 * 16) return;
  int e    = idx & 15;
  int lane = (idx >> 4) & 31;
  int nc   = (idx >> 9) & 1;
  int kc   = (idx >> 10) & 1;
  int g    = idx >> 11;
  int l = lane & 15, hi = lane >> 4;
  int K   = hi * 8 + (e < 8 ? e : e + 8);          // 0..31 within chunk
  int col = nc * 16 + l;
  float v;
  if (kc == 0) {
    v = (K < NB_FIN) ? C[(g * 2 + K) * NB_FOUT + col] : 0.0f;
  } else {
    const float* L = (g == 0) ? Lz : ((g == 1) ? Lr : Lh);
    v = L[(NB_FOUT + K) * NB_FOUT + col];
  }
  wB[idx] = (_Float16)v;
}

// prop[b,n,f] = selfnorm[n] * X[b,n,f,t]   (self-loop term; scatter adds the rest)
__global__ void k_propinit(const float* __restrict__ X, const float* __restrict__ selfnorm,
                           float* __restrict__ prop, int t) {
  int idx = blockIdx.x * blockDim.x + threadIdx.x;        // idx = (b*N+n)*2+f
  if (idx >= NB_B * NB_N * NB_FIN) return;
  int n = (idx >> 1) % NB_N;
  prop[idx] = selfnorm[n] * X[(long)idx * NB_T + t];
}

// Shared propagation for ALL three gates: prop[b,col] += norm[e] * X[b,row,:,t]
// 4 atomics per edge (2 batches x 2 input features) instead of 96.
__global__ void k_pscatter(const int* __restrict__ row, const int* __restrict__ col,
                           const float* __restrict__ norm, const float* __restrict__ X,
                           float* __restrict__ prop, int E, int t) {
  int e = blockIdx.x * blockDim.x + threadIdx.x;
  if (e >= E) return;
  int r = row[e], c = col[e];
  float nm = norm[e];
#pragma unroll
  for (int b = 0; b < NB_B; ++b) {
#pragma unroll
    for (int f = 0; f < NB_FIN; ++f) {
      long xi = (((long)b * NB_N + r) * NB_FIN + f) * NB_T + t;
      long pi = ((long)b * NB_N + c) * NB_FIN + f;
      atomic_add_f32(&prop[pi], nm * X[xi]);
    }
  }
}

// ---------------- GRU step: WMMA gate matmuls + elementwise update ----------------
// One wave owns a 16-node tile. Gate pre-activation = [prop(pad32) | H] @ Bfused + bias,
// done as 2 (K-chunks) x 2 (N-chunks) v_wmma_f32_16x16x32_f16 per gate.
__global__ void __launch_bounds__(GRU_WAVES * 32)
k_gru(const float* __restrict__ prop, float* __restrict__ H, float* __restrict__ out,
      const float* __restrict__ bfold, const _Float16* __restrict__ wB,
      const float* __restrict__ probs, int t) {
  __shared__ _Float16 lds[GRU_WAVES][16][32];   // per-wave R*H tile (f16)

  const int wave = threadIdx.x >> 5;
  const int lane = threadIdx.x & 31;
  const int tile = blockIdx.x * GRU_WAVES + wave;   // exact coverage: EXEC all-ones
  const int b    = tile / TILES_PER_B;
  const int tn   = (tile - b * TILES_PER_B) * 16;
  const int l    = lane & 15;
  const int hi   = lane >> 4;

  // --- A fragment, chunk 0: [prop0, prop1, 0, ...]  (16-bit A, 16x32 MxK)
  //     lanes 0-15: elements 0..7 -> K=0..7; only K=0,1 nonzero.
  //     lanes 16-31 cover K=8..15,24..31 -> all zero.
  const long baseP = ((long)b * NB_N + tn + l) * NB_FIN;
  float p0 = (hi == 0) ? prop[baseP + 0] : 0.0f;
  float p1 = (hi == 0) ? prop[baseP + 1] : 0.0f;
  v16h aP;
#pragma unroll
  for (int e = 0; e < 16; ++e) aP[e] = (_Float16)0.0f;
  aP[0] = (_Float16)p0;
  aP[1] = (_Float16)p1;

  // --- A fragment, chunk 1: H in A-layout (row M = lane%16,
  //     elements 0..7 -> K = hi*8+0..7, elements 8..15 -> K = 16+hi*8+0..7)
  const long baseA = ((long)b * NB_N + tn + l) * NB_FOUT;
  v16h aHm;
#pragma unroll
  for (int e = 0; e < 16; ++e) {
    int kk = hi * 8 + (e < 8 ? e : e + 8);
    aHm[e] = (_Float16)H[baseA + kk];
  }

  // --- B fragments: pre-swizzled, 32B aligned per lane ---
  const v16h* Wv = (const v16h*)wB;
#define WB_FRAG(g, kc, nc) Wv[((((g)*2 + (kc)) * 2 + (nc)) * 32) + lane]

  // --- fused biases in C/D layout: col = nc*16 + lane%16 ---
  float bz0 = bfold[l],               bz1 = bfold[16 + l];
  float br0 = bfold[NB_FOUT + l],     br1 = bfold[NB_FOUT + 16 + l];
  float bh0 = bfold[2 * NB_FOUT + l], bh1 = bfold[2 * NB_FOUT + 16 + l];

  v8f cz0, cz1, cr0, cr1;
#pragma unroll
  for (int v = 0; v < 8; ++v) { cz0[v] = bz0; cz1[v] = bz1; cr0[v] = br0; cr1[v] = br1; }

  // Z gate
  cz0 = __builtin_amdgcn_wmma_f32_16x16x32_f16(false, aP,  false, WB_FRAG(0,0,0), (short)0, cz0, false, false);
  cz0 = __builtin_amdgcn_wmma_f32_16x16x32_f16(false, aHm, false, WB_FRAG(0,1,0), (short)0, cz0, false, false);
  cz1 = __builtin_amdgcn_wmma_f32_16x16x32_f16(false, aP,  false, WB_FRAG(0,0,1), (short)0, cz1, false, false);
  cz1 = __builtin_amdgcn_wmma_f32_16x16x32_f16(false, aHm, false, WB_FRAG(0,1,1), (short)0, cz1, false, false);
  // R gate
  cr0 = __builtin_amdgcn_wmma_f32_16x16x32_f16(false, aP,  false, WB_FRAG(1,0,0), (short)0, cr0, false, false);
  cr0 = __builtin_amdgcn_wmma_f32_16x16x32_f16(false, aHm, false, WB_FRAG(1,1,0), (short)0, cr0, false, false);
  cr1 = __builtin_amdgcn_wmma_f32_16x16x32_f16(false, aP,  false, WB_FRAG(1,0,1), (short)0, cr1, false, false);
  cr1 = __builtin_amdgcn_wmma_f32_16x16x32_f16(false, aHm, false, WB_FRAG(1,1,1), (short)0, cr1, false, false);

  v8f Z0, Z1, R0, R1;
#pragma unroll
  for (int v = 0; v < 8; ++v) {
    Z0[v] = sigmoidf_(cz0[v]); Z1[v] = sigmoidf_(cz1[v]);
    R0[v] = sigmoidf_(cr0[v]); R1[v] = sigmoidf_(cr1[v]);
  }

  // --- H in C/D layout: row M = v + 8*hi, col = nc*16 + l ---
  float h0[8], h1[8];
#pragma unroll
  for (int v = 0; v < 8; ++v) {
    long ic = ((long)b * NB_N + tn + v + 8 * hi) * NB_FOUT + l;
    h0[v] = H[ic];
    h1[v] = H[ic + 16];
  }

  // --- transpose R*H from C-layout to A-layout through per-wave LDS tile ---
#pragma unroll
  for (int v = 0; v < 8; ++v) {
    lds[wave][v + 8 * hi][l]      = (_Float16)(R0[v] * h0[v]);
    lds[wave][v + 8 * hi][16 + l] = (_Float16)(R1[v] * h1[v]);
  }
  // DS ops are in-order per wave; force completion + block compiler reordering.
  asm volatile("s_wait_dscnt 0" ::: "memory");

  v16h aRH;
#pragma unroll
  for (int e = 0; e < 16; ++e) {
    int kk = hi * 8 + (e < 8 ? e : e + 8);
    aRH[e] = lds[wave][l][kk];
  }

  // H-candidate gate
  v8f ch0, ch1;
#pragma unroll
  for (int v = 0; v < 8; ++v) { ch0[v] = bh0; ch1[v] = bh1; }
  ch0 = __builtin_amdgcn_wmma_f32_16x16x32_f16(false, aP,  false, WB_FRAG(2,0,0), (short)0, ch0, false, false);
  ch0 = __builtin_amdgcn_wmma_f32_16x16x32_f16(false, aRH, false, WB_FRAG(2,1,0), (short)0, ch0, false, false);
  ch1 = __builtin_amdgcn_wmma_f32_16x16x32_f16(false, aP,  false, WB_FRAG(2,0,1), (short)0, ch1, false, false);
  ch1 = __builtin_amdgcn_wmma_f32_16x16x32_f16(false, aRH, false, WB_FRAG(2,1,1), (short)0, ch1, false, false);
#undef WB_FRAG

  const float p = probs[t];
#pragma unroll
  for (int v = 0; v < 8; ++v) {
    long ic = ((long)b * NB_N + tn + v + 8 * hi) * NB_FOUT + l;
    float ht0 = tanhf(ch0[v]);
    float ht1 = tanhf(ch1[v]);
    float hn0 = Z0[v] * h0[v] + (1.0f - Z0[v]) * ht0;
    float hn1 = Z1[v] * h1[v] + (1.0f - Z1[v]) * ht1;
    H[ic]        = hn0;
    H[ic + 16]   = hn1;
    out[ic]      += p * hn0;
    out[ic + 16] += p * hn1;
  }
}

// ---------------- host launcher ----------------
static inline size_t align256(size_t x) { return (x + 255) & ~(size_t)255; }

extern "C" void kernel_launch(void* const* d_in, const int* in_sizes, int n_in,
                              void* d_out, int out_size, void* d_ws, size_t ws_size,
                              hipStream_t stream) {
  (void)n_in; (void)ws_size;

  const float* X    = (const float*)d_in[0];
  const int*   ei   = (const int*)d_in[1];
  const float* ew   = (const float*)d_in[2];
  const float* Wz   = (const float*)d_in[3];
  const float* bz   = (const float*)d_in[4];
  const float* Wr   = (const float*)d_in[5];
  const float* br   = (const float*)d_in[6];
  const float* Wh   = (const float*)d_in[7];
  const float* bh   = (const float*)d_in[8];
  const float* Lz_w = (const float*)d_in[9];
  const float* Lz_b = (const float*)d_in[10];
  const float* Lr_w = (const float*)d_in[11];
  const float* Lr_b = (const float*)d_in[12];
  const float* Lh_w = (const float*)d_in[13];
  const float* Lh_b = (const float*)d_in[14];
  const float* att  = (const float*)d_in[15];

  const int E = in_sizes[2];            // edge_weight count
  const int* row = ei;                  // edge_index[0] = sources
  const int* col = ei + E;              // edge_index[1] = targets
  const long perH = (long)NB_B * NB_N * NB_FOUT;   // 3,200,000
  const long perP = (long)NB_B * NB_N * NB_FIN;    //   200,000

  // workspace carve
  char* w = (char*)d_ws;
  float*    probs = (float*)w;     w += align256(NB_T * sizeof(float));
  _Float16* wB    = (_Float16*)w;  w += align256((size_t)3 * 2 * 2 * 32 * 16 * sizeof(_Float16));
  float*    Cf    = (float*)w;     w += align256((size_t)3 * 2 * NB_FOUT * sizeof(float));
  float*    bfold = (float*)w;     w += align256((size_t)3 * NB_FOUT * sizeof(float));
  float*    deg   = (float*)w;     w += align256((size_t)NB_N * sizeof(float));
  float*    dinv  = (float*)w;     w += align256((size_t)NB_N * sizeof(float));
  float*    norm  = (float*)w;     w += align256(((size_t)E + NB_N) * sizeof(float));
  float*    prop  = (float*)w;     w += align256((size_t)perP * sizeof(float));
  float*    H     = (float*)w;     w += align256((size_t)perH * sizeof(float));
  float*    out   = (float*)d_out;

  const int TB = 256;

  // graph normalization + weight fusion (once)
  k_zero<<<(NB_N + TB - 1) / TB, TB, 0, stream>>>(deg, (long)NB_N);
  k_deg<<<(E + TB - 1) / TB, TB, 0, stream>>>(col, ew, deg, E);
  k_dinv<<<(NB_N + TB - 1) / TB, TB, 0, stream>>>(deg, dinv);
  k_norm<<<(E + NB_N + TB - 1) / TB, TB, 0, stream>>>(row, col, ew, dinv, norm, E);
  k_softmax<<<1, 32, 0, stream>>>(att, probs);
  k_fuse<<<1, 128, 0, stream>>>(Wz, bz, Wr, br, Wh, bh,
                                Lz_w, Lz_b, Lr_w, Lr_b, Lh_w, Lh_b, Cf, bfold);
  k_prepw<<<(3 * 2 * 2 * 32 * 16 + TB - 1) / TB, TB, 0, stream>>>(Lz_w, Lr_w, Lh_w, Cf, wB);

  // state init
  k_zero<<<(int)((perH + TB - 1) / TB), TB, 0, stream>>>(H, perH);
  k_zero<<<(out_size + TB - 1) / TB, TB, 0, stream>>>(out, (long)out_size);

  for (int t = 0; t < NB_T; ++t) {
    k_propinit<<<(int)((perP + TB - 1) / TB), TB, 0, stream>>>(X, norm + E, prop, t);
    k_pscatter<<<(E + TB - 1) / TB, TB, 0, stream>>>(row, col, norm, X, prop, E, t);
    k_gru<<<(NB_B * TILES_PER_B) / GRU_WAVES, GRU_WAVES * 32, 0, stream>>>(
        prop, H, out, bfold, wB, probs, t);
  }
}
